// SageConvLayer_10582799417849
// MI455X (gfx1250) — compile-verified, matrix operand order checked by reading, not run on hardware
//
#include <hip/hip_runtime.h>

// ---------------------------------------------------------------------------
// SAGE conv for MI455X (gfx1250): bf16 WMMA (v_wmma_f32_16x16x32_bf16),
// fp32 accumulate. adj (256 MB) is streamed exactly once via async
// global->LDS copies (ASYNCcnt, double-buffered staging); rowsum/degree
// normalization is fused into the aggregation GEMM via LDS float atomics.
// ---------------------------------------------------------------------------

typedef __attribute__((ext_vector_type(16))) __bf16 bf16x16;  // 8 VGPRs
typedef __attribute__((ext_vector_type(8)))  float  f32x8;    // 8 VGPRs

constexpr int NN   = 8192;  // nodes
constexpr int FF   = 256;   // feature dim
constexpr int OUTF = 256;   // output dim
constexpr int TWOF = 512;   // concat dim

// Load one 16x32 bf16 A/B fragment row for this lane.
// rowptr points at element 0 of this lane's row; koff = k_base + (hi?8:0).
// Per ISA layout: lanes 0-15 hold K[0..7] (VGPR0-3) and K[16..23] (VGPR4-7);
// lanes 16-31 hold K[8..15] and K[24..31]. Two 16-byte loads.
__device__ __forceinline__ bf16x16 load_frag(const __bf16* rowptr, int koff) {
    union { uint4 u[2]; bf16x16 v; } f;
    const uint4* p = reinterpret_cast<const uint4*>(rowptr + koff);
    f.u[0] = p[0];   // K koff .. koff+7
    f.u[1] = p[2];   // K koff+16 .. koff+23  (+32 bytes)
    return f.v;
}

__device__ __forceinline__ bf16x16 ld_frag_ld(const __bf16* rowptr, int koff) {
    // same pattern but from LDS (ds_load_b128 x2)
    union { uint4 u[2]; bf16x16 v; } f;
    const uint4* p = reinterpret_cast<const uint4*>(rowptr + koff);
    f.u[0] = p[0];
    f.u[1] = p[2];
    return f.v;
}

#define WMMA_BF16(A, B, C) \
    __builtin_amdgcn_wmma_f32_16x16x32_bf16(false, (A), false, (B), (short)0, (C), false, false)

// Async global->LDS copy of 16 bytes (GLOBAL_LOAD_ASYNC_TO_LDS_B128, GV mode).
// LDS dest offset in a VGPR, 64-bit global address in a VGPR pair. ASYNCcnt.
__device__ __forceinline__ void async_to_lds_b128(void* lds_dst, const void* gsrc) {
    unsigned loff = (unsigned)(size_t)lds_dst;               // low 32b = LDS offset
    unsigned long long ga = (unsigned long long)(size_t)gsrc;
    asm volatile("global_load_async_to_lds_b128 %0, %1, off"
                 :: "v"(loff), "v"(ga)
                 : "memory");
}

// ------------------------- fp32 -> bf16 conversion -------------------------
__global__ void cvt_f32_bf16_kernel(const float* __restrict__ src,
                                    __bf16* __restrict__ dst, int n) {
    int i = blockIdx.x * blockDim.x + threadIdx.x;
    int stride = gridDim.x * blockDim.x;
    for (; i < n; i += stride) dst[i] = (__bf16)src[i];
}

// --------------- GEMM1: hT[f][node] = (feat @ Wn^T)^T, bf16 ----------------
// Block: 256 threads = 8 waves. Block tile: 64 rows x 256 cols.
// Wave (w): rows m0..m0+15 (m0 = blk*64 + (w/2)*16), cols n0..n0+127 (n0=(w&1)*128).
__global__ __launch_bounds__(256, 2)
void gemm_neigh_hT(const __bf16* __restrict__ featB,
                   const __bf16* __restrict__ Wn,   // [FF][FF], row j = B column j
                   __bf16* __restrict__ hT) {       // [FF][NN]
    const int lane = threadIdx.x & 31;
    const int wave = threadIdx.x >> 5;
    const int l    = lane & 15;
    const int hi   = (lane >> 4) & 1;
    const int m0   = blockIdx.x * 64 + (wave >> 1) * 16;
    const int n0   = (wave & 1) * 128;

    f32x8 acc[8] = {};
    const __bf16* arow = featB + (m0 + l) * FF;

    for (int kb = 0; kb < FF; kb += 32) {
        bf16x16 a = load_frag(arow, kb + hi * 8);
#pragma unroll
        for (int t = 0; t < 8; ++t) {
            bf16x16 b = load_frag(Wn + (n0 + t * 16 + l) * FF, kb + hi * 8);
            acc[t] = WMMA_BF16(a, b, acc[t]);
        }
    }
#pragma unroll
    for (int t = 0; t < 8; ++t)
#pragma unroll
        for (int v = 0; v < 8; ++v)
            hT[(n0 + t * 16 + l) * NN + (m0 + v + hi * 8)] = (__bf16)acc[t][v];
}

// ------ GEMM2 (fused): hagg = (adj @ h) / (rowsum(adj)+1), bf16 output ------
// adj streamed once. Each 64x32 fp32 tile is fetched with async global->LDS
// copies into a double-buffered staging area (no VGPR round-trip), then
// converted in place to a bf16 WMMA tile; partial row sums via ds_add_f32.
__global__ __launch_bounds__(256, 2)
void gemm_agg(const float* __restrict__ adj,   // [NN][NN] fp32
              const __bf16* __restrict__ hT,   // [FF][NN] bf16 (B columns contiguous)
              __bf16* __restrict__ hagg) {     // [NN][FF] bf16
    __shared__ float  Af[2][64 * 32];   // double-buffered fp32 staging (16 KB)
    __shared__ __bf16 As[64 * 32];      // bf16 WMMA tile (4 KB)
    __shared__ float  rowsum[64];

    const int tid  = threadIdx.x;
    const int lane = tid & 31;
    const int wave = tid >> 5;
    const int l    = lane & 15;
    const int hi   = (lane >> 4) & 1;
    const int mblk = blockIdx.x * 64;
    const int msub = (wave >> 1) * 16;
    const int n0   = (wave & 1) * 128;

    if (tid < 64) rowsum[tid] = 0.0f;

    // adj tile ownership: 4 threads per row, 8 fp32 (32 bytes) each. Each
    // thread's staging slice is written (async) and read (convert) only by
    // itself, so per-wave s_wait_asynccnt is the only ordering needed there.
    const int r  = tid >> 2;          // 0..63
    const int c0 = (tid & 3) * 8;     // 0,8,16,24
    const float* arow = adj + (size_t)(mblk + r) * NN + c0;
    float* stage0 = &Af[0][r * 32 + c0];
    float* stage1 = &Af[1][r * 32 + c0];

    // Prologue: kick off tile 0 into buffer 0.
    async_to_lds_b128(stage0,     arow);
    async_to_lds_b128(stage0 + 4, arow + 4);

    f32x8 acc[8] = {};
    __syncthreads();   // rowsum init visible

    int buf = 0;
    for (int kk = 0; kk < NN; kk += 32, buf ^= 1) {
        float* cur = buf ? stage1 : stage0;
        float* nxt = buf ? stage0 : stage1;
        if (kk + 32 < NN) {
            // Prefetch next tile into the other buffer, then wait only for
            // the current tile (2 newest async ops stay in flight).
            async_to_lds_b128(nxt,     arow + kk + 32);
            async_to_lds_b128(nxt + 4, arow + kk + 36);
            asm volatile("s_wait_asynccnt 0x2" ::: "memory");
        } else {
            asm volatile("s_wait_asynccnt 0x0" ::: "memory");
        }

        // Convert own fp32 slice -> bf16 tile + accumulate row sum.
        float4 f0 = *reinterpret_cast<const float4*>(cur);
        float4 f1 = *reinterpret_cast<const float4*>(cur + 4);
        atomicAdd(&rowsum[r], f0.x + f0.y + f0.z + f0.w + f1.x + f1.y + f1.z + f1.w);
        __bf16* dstp = &As[r * 32 + c0];
        dstp[0] = (__bf16)f0.x; dstp[1] = (__bf16)f0.y;
        dstp[2] = (__bf16)f0.z; dstp[3] = (__bf16)f0.w;
        dstp[4] = (__bf16)f1.x; dstp[5] = (__bf16)f1.y;
        dstp[6] = (__bf16)f1.z; dstp[7] = (__bf16)f1.w;
        __syncthreads();

        bf16x16 a = ld_frag_ld(&As[(msub + l) * 32], hi * 8);
#pragma unroll
        for (int t = 0; t < 8; ++t) {
            bf16x16 b = load_frag(hT + (size_t)(n0 + t * 16 + l) * NN, kk + hi * 8);
            acc[t] = WMMA_BF16(a, b, acc[t]);
        }
        __syncthreads();   // before next iteration overwrites As
    }

#pragma unroll
    for (int v = 0; v < 8; ++v) {
        const int ml  = msub + v + hi * 8;
        const float s = 1.0f / (rowsum[ml] + 1.0f);
#pragma unroll
        for (int t = 0; t < 8; ++t)
            hagg[(size_t)(mblk + ml) * FF + n0 + t * 16 + l] = (__bf16)(acc[t][v] * s);
    }
}

// --------- GEMM3: z = feat_bf16 @ W1^T + hagg @ W2^T, fp32 output ----------
__global__ __launch_bounds__(256, 2)
void gemm_out(const __bf16* __restrict__ featB,
              const __bf16* __restrict__ hagg,
              const __bf16* __restrict__ Wl,   // [OUTF][TWOF], row o = B column o
              float* __restrict__ z) {         // [NN][OUTF] fp32
    const int lane = threadIdx.x & 31;
    const int wave = threadIdx.x >> 5;
    const int l    = lane & 15;
    const int hi   = (lane >> 4) & 1;
    const int m0   = blockIdx.x * 64 + (wave >> 1) * 16;
    const int n0   = (wave & 1) * 128;

    f32x8 acc[8] = {};
    const __bf16* a1row = featB + (m0 + l) * FF;
    const __bf16* a2row = hagg  + (m0 + l) * FF;

    for (int kb = 0; kb < FF; kb += 32) {
        bf16x16 a1 = load_frag(a1row, kb + hi * 8);
        bf16x16 a2 = load_frag(a2row, kb + hi * 8);
#pragma unroll
        for (int t = 0; t < 8; ++t) {
            const __bf16* wrow = Wl + (size_t)(n0 + t * 16 + l) * TWOF;
            bf16x16 b1 = load_frag(wrow,      kb + hi * 8);   // W1 part (k < 256)
            bf16x16 b2 = load_frag(wrow + FF, kb + hi * 8);   // W2 part (k >= 256)
            acc[t] = WMMA_BF16(a1, b1, acc[t]);
            acc[t] = WMMA_BF16(a2, b2, acc[t]);
        }
    }
#pragma unroll
    for (int t = 0; t < 8; ++t)
#pragma unroll
        for (int v = 0; v < 8; ++v)
            z[(m0 + v + hi * 8) * OUTF + n0 + t * 16 + l] = acc[t][v];
}

// ---------------------------------------------------------------------------
extern "C" void kernel_launch(void* const* d_in, const int* in_sizes, int n_in,
                              void* d_out, int out_size, void* d_ws, size_t ws_size,
                              hipStream_t stream) {
    const float* features = (const float*)d_in[0];   // [NN][FF]
    const float* adj      = (const float*)d_in[1];   // [NN][NN]
    const float* neigh_W  = (const float*)d_in[2];   // [FF][FF]
    const float* lin_W    = (const float*)d_in[3];   // [OUTF][TWOF]
    float* z = (float*)d_out;

    // Workspace carve-up (~12.5 MB total)
    __bf16* featB = reinterpret_cast<__bf16*>(d_ws);          // NN*FF   = 4 MB
    __bf16* Wn    = featB + (size_t)NN * FF;                  // FF*FF   = 128 KB
    __bf16* Wl    = Wn + (size_t)FF * FF;                     // OUTF*TWOF = 256 KB
    __bf16* hT    = Wl + (size_t)OUTF * TWOF;                 // FF*NN   = 4 MB
    __bf16* hagg  = hT + (size_t)FF * NN;                     // NN*FF   = 4 MB

    auto blocks = [](int n) { int b = (n + 255) / 256; return b > 2048 ? 2048 : b; };
    cvt_f32_bf16_kernel<<<blocks(NN * FF), 256, 0, stream>>>(features, featB, NN * FF);
    cvt_f32_bf16_kernel<<<blocks(FF * FF), 256, 0, stream>>>(neigh_W, Wn, FF * FF);
    cvt_f32_bf16_kernel<<<blocks(OUTF * TWOF), 256, 0, stream>>>(lin_W, Wl, OUTF * TWOF);

    gemm_neigh_hT<<<NN / 64, 256, 0, stream>>>(featB, Wn, hT);
    gemm_agg<<<NN / 64, 256, 0, stream>>>(adj, hT, hagg);
    gemm_out<<<NN / 64, 256, 0, stream>>>(featB, hagg, Wl, z);
}
